// CrossAttnLayer_76957224010028
// MI455X (gfx1250) — matrix operand order repaired and checked by reading, not compiled, not run on hardware
//
#include <hip/hip_runtime.h>
#include <hip/hip_bf16.h>
#include <math.h>

// ---------------------------------------------------------------------------
// GAT layer for MI455X (gfx1250, wave32).
//   proj = x @ W  via v_wmma_f32_16x16x32_f16, W^T staged in LDS (ds_load_b128)
//   scores, global-max softmax shift, edge exp + segment-sum (f32 atomics),
//   att normalization pass, feature scatter (L2-resident gather +
//   global_atomic_add_f32), skip + bias + ELU.
// ---------------------------------------------------------------------------

#define FIN   128
#define NHEAD 4
#define FOUT  32
#define SLOPE 0.2f

typedef __attribute__((ext_vector_type(16))) _Float16 v16h;
typedef __attribute__((ext_vector_type(8)))  _Float16 v8h;
typedef __attribute__((ext_vector_type(8)))  float    v8f;

// ---- ordered-uint mapping for float atomicMax --------------------------------
__device__ __forceinline__ unsigned f2ord(float f) {
  unsigned u = __float_as_uint(f);
  return (u & 0x80000000u) ? ~u : (u | 0x80000000u);
}
__device__ __forceinline__ float ord2f(unsigned u) {
  u = (u & 0x80000000u) ? (u & 0x7fffffffu) : ~u;
  return __uint_as_float(u);
}

// ---- 0. per-launch init: zero accumulators, seed global max -------------------
__global__ void gat_init(float* __restrict__ out, float* __restrict__ denom,
                         unsigned* __restrict__ gmax, int n_out, int n_den) {
  int gid = blockIdx.x * blockDim.x + threadIdx.x;
  if (gid < n_out) out[gid] = 0.0f;
  if (gid < n_den) denom[gid] = 0.0f;
  if (gid == 0)    *gmax = f2ord(-INFINITY);
}

// ---- 1a. convert x -> f16 -----------------------------------------------------
__global__ void gat_cvt_x(const float* __restrict__ x, _Float16* __restrict__ xb,
                          int total) {
  int gid = blockIdx.x * blockDim.x + threadIdx.x;
  if (gid < total) xb[gid] = (_Float16)x[gid];
}

// ---- 1b. convert W -> f16, transposed (wt[n][k] = W[k][n]) --------------------
__global__ void gat_cvt_w(const float* __restrict__ W, _Float16* __restrict__ wt) {
  int gid = blockIdx.x * blockDim.x + threadIdx.x;   // 128*128
  if (gid < FIN * FIN) {
    int k = gid >> 7, n = gid & 127;
    wt[n * FIN + k] = (_Float16)W[k * FIN + n];
  }
}

// ---- 2. proj = x @ W via WMMA. One wave -> 16 rows x 128 cols -----------------
// W^T (32KB) is staged in LDS once per 4-wave workgroup; each K-chunk loads all
// 8 B fragments (ds_load_b128) before issuing 8 accumulating WMMAs back-to-back.
__global__ __launch_bounds__(128) void gat_gemm(
    const _Float16* __restrict__ xb,   // [N][128] f16
    const _Float16* __restrict__ wt,   // [128 cols][128 k] f16 (transposed W)
    float* __restrict__ proj,          // [N][128] f32
    int N, int ntiles) {
  __shared__ _Float16 wls[FIN * FIN];                 // 32 KB of 320 KB/WGP

  // cooperative fill: 128 threads x 16 x 16B = 32KB
  const int tid = threadIdx.x;
  #pragma unroll
  for (int i = 0; i < 16; ++i) {
    const int idx = (i * 128 + tid) * 8;              // half index
    *reinterpret_cast<v8h*>(&wls[idx]) = *reinterpret_cast<const v8h*>(wt + idx);
  }
  __syncthreads();

  const int lane = tid & 31;
  const int wave = tid >> 5;
  const int tile = blockIdx.x * 4 + wave;
  if (tile >= ntiles) return;                 // uniform per wave: EXEC stays all-1s
  const int m0 = tile * 16;

  // A fragment addressing (contiguous k order: element s of lane half h is k = 16h + s)
  int arow = m0 + (lane & 15);
  if (arow >= N) arow = N - 1;                // tail clamp (N=50000 is exact anyway)
  const int akb = (lane >> 4) * 16;           // K sub-base: 0 or 16
  // B fragment addressing: lane covers col (lane&15), K run of 16
  const int bn  = lane & 15;
  const int bkb = (lane >> 4) * 16;

  v8f acc[8] = {};                            // 8 col-tiles of 16x16 f32
  #pragma unroll
  for (int kc = 0; kc < 4; ++kc) {
    const int k0 = kc * 32;
    v8h alo = *reinterpret_cast<const v8h*>(xb + (size_t)arow * FIN + k0 + akb);
    v8h ahi = *reinterpret_cast<const v8h*>(xb + (size_t)arow * FIN + k0 + akb + 8);
    v16h a;
    #pragma unroll
    for (int i = 0; i < 8; ++i) { a[i] = alo[i]; a[i + 8] = ahi[i]; }

    v16h b[8];
    #pragma unroll
    for (int nt = 0; nt < 8; ++nt) {
      const int n = nt * 16 + bn;
      b[nt] = *reinterpret_cast<const v16h*>(&wls[n * FIN + k0 + bkb]);
    }
    #pragma unroll
    for (int nt = 0; nt < 8; ++nt) {
      acc[nt] = __builtin_amdgcn_wmma_f32_16x16x32_f16(
          false, a, false, b[nt], (short)0, acc[nt], false, false);
    }
  }
  // C/D layout: VGPR r -> row r + 8*(lane>>4), col = lane&15
  const int rbase = (lane >> 4) * 8;
  const int col   = lane & 15;
  #pragma unroll
  for (int nt = 0; nt < 8; ++nt) {
    #pragma unroll
    for (int r = 0; r < 8; ++r) {
      const int row = m0 + rbase + r;
      if (row < N) proj[(size_t)row * FIN + nt * 16 + col] = acc[nt][r];
    }
  }
}

// ---- 3. per-node attention scores: s = <proj[n,h,:], a_{src,trg}[h,:]> --------
__global__ void gat_scores(const float* __restrict__ proj,
                           const float* __restrict__ a_src,
                           const float* __restrict__ a_trg,
                           float* __restrict__ s_src, float* __restrict__ s_trg,
                           int N) {
  int gid = blockIdx.x * blockDim.x + threadIdx.x;   // (n, h)
  if (gid >= N * NHEAD) return;
  const int n = gid >> 2, h = gid & 3;
  const float* p = proj + (size_t)n * FIN + h * FOUT;
  float ss = 0.f, st = 0.f;
  #pragma unroll 8
  for (int f = 0; f < FOUT; ++f) {
    const float v = p[f];
    ss += v * a_src[h * FOUT + f];
    st += v * a_trg[h * FOUT + f];
  }
  s_src[gid] = ss;
  s_trg[gid] = st;
}

// ---- 4. global max of leaky_relu edge scores ----------------------------------
__global__ void gat_edge_max(const long long* __restrict__ ei,
                             const float* __restrict__ s_src,
                             const float* __restrict__ s_trg,
                             unsigned* __restrict__ gmax, int E) {
  __shared__ float sm[256];
  int gid = blockIdx.x * blockDim.x + threadIdx.x;   // (e, h)
  float v = -INFINITY;
  if (gid < E * NHEAD) {
    const int e = gid >> 2, h = gid & 3;
    const long long s = ei[e], t = ei[E + e];
    float x = s_src[s * NHEAD + h] + s_trg[t * NHEAD + h];
    v = x > 0.f ? x : SLOPE * x;
  }
  sm[threadIdx.x] = v;
  __syncthreads();
  for (int o = 128; o > 0; o >>= 1) {
    if (threadIdx.x < o) sm[threadIdx.x] = fmaxf(sm[threadIdx.x], sm[threadIdx.x + o]);
    __syncthreads();
  }
  if (threadIdx.x == 0) atomicMax(gmax, f2ord(sm[0]));
}

// ---- 5. exp(e - max), segment-sum denom by target -----------------------------
__global__ void gat_edge_exp(const long long* __restrict__ ei,
                             const float* __restrict__ s_src,
                             const float* __restrict__ s_trg,
                             const unsigned* __restrict__ gmaxp,
                             float* __restrict__ exp_e,
                             float* __restrict__ denom, int E) {
  int gid = blockIdx.x * blockDim.x + threadIdx.x;   // (e, h)
  if (gid >= E * NHEAD) return;
  const int e = gid >> 2, h = gid & 3;
  const long long s = ei[e], t = ei[E + e];
  float x = s_src[s * NHEAD + h] + s_trg[t * NHEAD + h];
  x = x > 0.f ? x : SLOPE * x;
  const float gm = ord2f(*gmaxp);
  const float w = __expf(x - gm);
  exp_e[gid] = w;
  unsafeAtomicAdd(&denom[t * NHEAD + h], w);         // global_atomic_add_f32
}

// ---- 5b. normalize: att = exp_e / (denom[trg] + eps), in place ----------------
// Hoists the divide out of the 102M-thread scatter hot loop.
__global__ void gat_att(const long long* __restrict__ ei,
                        float* __restrict__ exp_e,
                        const float* __restrict__ denom, int E) {
  int gid = blockIdx.x * blockDim.x + threadIdx.x;   // (e, h)
  if (gid >= E * NHEAD) return;
  const int e = gid >> 2, h = gid & 3;
  const long long t = ei[E + e];
  exp_e[gid] = exp_e[gid] / (denom[t * NHEAD + h] + 1e-16f);
}

// ---- 6. feature scatter: out[trg] += proj[src] * att --------------------------
// One thread per (edge, feature): 512B coalesced gather per edge; proj and out
// both fit in 192MB L2 -> L2-resident RMW traffic. Pure mult + atomic fadd.
__global__ void gat_scatter(const long long* __restrict__ ei,
                            const float* __restrict__ proj,
                            const float* __restrict__ att,
                            float* __restrict__ out, int E) {
  const unsigned gid = blockIdx.x * 256u + threadIdx.x;
  const int e = gid >> 7;
  if (e >= E) return;
  const int c = gid & 127, h = c >> 5;
  const long long s = ei[e], t = ei[E + e];
  unsafeAtomicAdd(&out[t * (size_t)FIN + c],
                  proj[s * (size_t)FIN + c] * att[e * NHEAD + h]);
}

// ---- 7. skip connection + bias + ELU (in place on d_out) ----------------------
__global__ void gat_finalize(const float* __restrict__ x,
                             const float* __restrict__ bias,
                             float* __restrict__ out, int total) {
  int gid = blockIdx.x * blockDim.x + threadIdx.x;
  if (gid >= total) return;
  const float v = out[gid] + x[gid] + bias[gid & 127];
  out[gid] = v > 0.f ? v : expm1f(v);
}

// ---------------------------------------------------------------------------
extern "C" void kernel_launch(void* const* d_in, const int* in_sizes, int n_in,
                              void* d_out, int out_size, void* d_ws, size_t ws_size,
                              hipStream_t stream) {
  const float*     x    = (const float*)d_in[0];
  const long long* ei   = (const long long*)d_in[1];   // int64 edge_index (2,E)
  const float*     W    = (const float*)d_in[2];
  const float*     asrc = (const float*)d_in[3];
  const float*     atrg = (const float*)d_in[4];
  const float*     bias = (const float*)d_in[5];
  float*           out  = (float*)d_out;

  const int N = in_sizes[0] / FIN;        // 50000
  const int E = in_sizes[1] / 2;          // 800000

  // workspace layout
  char* ws = (char*)d_ws;
  _Float16* xb   = (_Float16*)ws;  ws += (size_t)N * FIN * sizeof(_Float16);
  _Float16* wt   = (_Float16*)ws;  ws += (size_t)FIN * FIN * sizeof(_Float16);
  float*    proj = (float*)ws;     ws += (size_t)N * FIN * sizeof(float);
  float*    ssrc = (float*)ws;     ws += (size_t)N * NHEAD * sizeof(float);
  float*    strg = (float*)ws;     ws += (size_t)N * NHEAD * sizeof(float);
  float*    den  = (float*)ws;     ws += (size_t)N * NHEAD * sizeof(float);
  float*    expe = (float*)ws;     ws += (size_t)E * NHEAD * sizeof(float);
  unsigned* gmax = (unsigned*)ws;

  const int nOut = N * FIN;               // 6.4M
  const int nDen = N * NHEAD;             // 200K
  const int nEH  = E * NHEAD;             // 3.2M
  const int B = 256;

  gat_init<<<(nOut + B - 1) / B, B, 0, stream>>>(out, den, gmax, nOut, nDen);
  gat_cvt_x<<<(nOut + B - 1) / B, B, 0, stream>>>(x, xb, nOut);
  gat_cvt_w<<<(FIN * FIN + B - 1) / B, B, 0, stream>>>(W, wt);

  const int ntiles = (N + 15) / 16;       // 3125
  gat_gemm<<<(ntiles + 3) / 4, 128, 0, stream>>>(xb, wt, proj, N, ntiles);

  gat_scores<<<(nDen + B - 1) / B, B, 0, stream>>>(proj, asrc, atrg, ssrc, strg, N);
  gat_edge_max<<<(nEH + B - 1) / B, B, 0, stream>>>(ei, ssrc, strg, gmax, E);
  gat_edge_exp<<<(nEH + B - 1) / B, B, 0, stream>>>(ei, ssrc, strg, gmax, expe, den, E);
  gat_att<<<(nEH + B - 1) / B, B, 0, stream>>>(ei, expe, den, E);

  const unsigned scatterBlocks = (unsigned)(((size_t)E * FIN + B - 1) / B); // 400000
  gat_scatter<<<scatterBlocks, B, 0, stream>>>(ei, proj, expe, out, E);

  gat_finalize<<<(nOut + B - 1) / B, B, 0, stream>>>(x, bias, out, nOut);
}
